// RelationModuleMultiScale_936302870703
// MI455X (gfx1250) — matrix operand order repaired and checked by reading, not compiled, not run on hardware
//
#include <hip/hip_runtime.h>
#include <stdint.h>

// ---------------- problem constants ----------------
#define BATCH   16384
#define TFR     8
#define DDIM    256
#define FMAX    (TFR * DDIM)     // 2048, padded in-features of W1
#define KDIM    256              // bottleneck
#define CDIM    174              // classes
#define CPAD    192              // padded to 12 * 16 for WMMA N tiling
#define NREL    19
#define NSCALE  7

typedef __attribute__((ext_vector_type(16))) __bf16 v16bf;
typedef __attribute__((ext_vector_type(4)))  __bf16 v4bf;
typedef __attribute__((ext_vector_type(8)))  float  v8f;
typedef __attribute__((ext_vector_type(4)))  int    v4i;

// CDNA5 async global->LDS path (ASYNCcnt), with safe fallback
#if defined(__AMDGCN__) && __has_builtin(__builtin_amdgcn_global_load_async_to_lds_b128) && __has_builtin(__builtin_amdgcn_s_wait_asynccnt)
#define GASYNC_OK 1
#else
#define GASYNC_OK 0
#endif

#if GASYNC_OK
typedef __attribute__((address_space(1))) v4i* gas_gptr;
typedef __attribute__((address_space(3))) v4i* gas_lptr;
#endif

__device__ __forceinline__ void cp16(const __bf16* g, __bf16* l) {
#if GASYNC_OK
    gas_gptr gp = (gas_gptr)(uintptr_t)g;                  // global ptr -> AS1 int4*
    gas_lptr lp = (gas_lptr)(uint32_t)(uintptr_t)l;        // generic LDS ptr: low 32 bits = LDS offset
    __builtin_amdgcn_global_load_async_to_lds_b128(gp, lp, 0, 0);
#else
    *reinterpret_cast<uint4*>(l) = *reinterpret_cast<const uint4*>(g);
#endif
}

__device__ __forceinline__ void wait_async() {
#if GASYNC_OK
    __builtin_amdgcn_s_wait_asynccnt(0);
#endif
}

// relation table: scale index, #frames, frame ids (fixed deterministic stand-in
// for the module's seeded subsampling; structure identical to the reference)
__device__ __constant__ int c_rel_scale[NREL] =
    {0, 1,1,1, 2,2,2, 3,3,3, 4,4,4, 5,5,5, 6,6,6};
__device__ __constant__ int c_rel_s[NREL] =
    {8, 7,7,7, 6,6,6, 5,5,5, 4,4,4, 3,3,3, 2,2,2};
__device__ __constant__ int c_rel_frames[NREL][8] = {
    {0,1,2,3,4,5,6,7},
    {0,1,2,4,5,6,7,0},{0,1,2,3,4,6,7,0},{0,1,3,4,5,6,7,0},
    {0,1,3,4,6,7,0,0},{0,2,3,5,6,7,0,0},{1,2,3,4,5,7,0,0},
    {0,2,4,5,7,0,0,0},{1,2,3,6,7,0,0,0},{0,1,4,5,6,0,0,0},
    {0,3,5,7,0,0,0,0},{1,2,4,6,0,0,0,0},{0,2,5,6,0,0,0,0},
    {1,4,7,0,0,0,0,0},{0,3,6,0,0,0,0,0},{2,5,7,0,0,0,0,0},
    {0,7,0,0,0,0,0,0},{2,5,0,0,0,0,0,0},{3,6,0,0,0,0,0,0}};

// ---------------- prep kernels ----------------

__global__ void k_relu_cvt_x(const float* __restrict__ x,
                             __bf16* __restrict__ xr, int n4) {
    int i = blockIdx.x * blockDim.x + threadIdx.x;
    if (i >= n4) return;
    float4 v = reinterpret_cast<const float4*>(x)[i];
    v4bf o;
    o[0] = (__bf16)fmaxf(v.x, 0.f);
    o[1] = (__bf16)fmaxf(v.y, 0.f);
    o[2] = (__bf16)fmaxf(v.z, 0.f);
    o[3] = (__bf16)fmaxf(v.w, 0.f);
    reinterpret_cast<v4bf*>(xr)[i] = o;
}

// W1[i][f][n] (fp32) -> W1T[i][n][f] (bf16), n-major so B-fragments are K-contiguous
__global__ void k_cvt_w1t(const float* __restrict__ w1,
                          __bf16* __restrict__ w1t, int n) {
    int idx = blockIdx.x * blockDim.x + threadIdx.x;
    if (idx >= n) return;
    int i = idx / (KDIM * FMAX);
    int r = idx % (KDIM * FMAX);
    int nn = r / FMAX;
    int f  = r % FMAX;
    w1t[idx] = (__bf16)w1[((size_t)i * FMAX + f) * KDIM + nn];
}

// W2[i][k][c] (fp32) -> W2T[i][c_pad192][k] (bf16), zero pad c>=174
__global__ void k_cvt_w2t(const float* __restrict__ w2,
                          __bf16* __restrict__ w2t, int n) {
    int idx = blockIdx.x * blockDim.x + threadIdx.x;
    if (idx >= n) return;
    int i = idx / (CPAD * KDIM);
    int r = idx % (CPAD * KDIM);
    int c = r / KDIM;
    int k = r % KDIM;
    float v = (c < CDIM) ? w2[((size_t)i * KDIM + k) * CDIM + c] : 0.f;
    w2t[idx] = (__bf16)v;
}

// out[b][c] = b2[0][c] + 3 * sum_{i=1..6} b2[i][c]   (bias seed; GEMMs atomically add)
__global__ void k_init_out(const float* __restrict__ b2,
                           float* __restrict__ out, int n) {
    int idx = blockIdx.x * blockDim.x + threadIdx.x;
    if (idx >= n) return;
    int c = idx % CDIM;
    float v = b2[c];
    #pragma unroll
    for (int i = 1; i < NSCALE; ++i) v += 3.f * b2[i * CDIM + c];
    out[idx] = v;
}

// ---------------- fused main kernel ----------------
// grid: (BATCH/64, NREL), block: 256 (8 waves of 32)
// Stage1: H = relu(relu(G) @ W1 + b1)  (64 x 256 tile, K = s*256), double-buffered LDS
// Stage2: out += H @ W2                (64 x 192pad tile, K = 256)
__launch_bounds__(256, 1)
__global__ void k_trn_fused(const __bf16* __restrict__ xr,
                            const __bf16* __restrict__ w1t,
                            const __bf16* __restrict__ w2t,
                            const float*  __restrict__ b1,
                            float* __restrict__ out)
{
    __shared__ __align__(16) __bf16 lds_a[2][64][40];   // 64x32 A tiles, 80B pitch
    __shared__ __align__(16) __bf16 lds_b[2][256][40];  // 256(n) x 32(k) W1 tiles
    __shared__ __align__(16) __bf16 lds_h[64][264];     // 64 x 256 H tile, 528B pitch
    __shared__ __align__(16) __bf16 lds_w2[192][40];    // 192(c) x 32(k) W2 tile

    const int tid  = threadIdx.x;
    const int lane = tid & 31;
    const int w    = tid >> 5;
    const int half = lane >> 4;      // 0: lanes 0-15, 1: lanes 16-31
    const int l16  = lane & 15;

    const int rel = blockIdx.y;
    const int isc = c_rel_scale[rel];
    const int s   = c_rel_s[rel];
    const int m0  = blockIdx.x * 64;

    // ---- stage 1: wave (wm in 0..1 over M, wn in 0..3 over N) ----
    const int wm = w & 1;
    const int wn = w >> 1;

    v8f acc[2][4];
    #pragma unroll
    for (int mm = 0; mm < 2; ++mm)
        #pragma unroll
        for (int nn = 0; nn < 4; ++nn)
            #pragma unroll
            for (int e = 0; e < 8; ++e) acc[mm][nn][e] = 0.f;

    // tile copy roles
    const int arow = tid >> 2, ach = tid & 3;           // A: 64 rows x 4 x 16B
    const __bf16* gb_base = w1t + (size_t)(isc * KDIM + tid) * FMAX;

    auto issue_tile = [&](int ks, int buf) {
        const int kk    = ks * 32;
        const int frame = c_rel_frames[rel][kk >> 8];
        const int koff  = kk & 255;
        const __bf16* ga = xr + (size_t)(m0 + arow) * (TFR * DDIM)
                              + frame * DDIM + koff + ach * 8;
        cp16(ga, &lds_a[buf][arow][ach * 8]);
        const __bf16* gb = gb_base + kk;
        cp16(gb +  0, &lds_b[buf][tid][ 0]);
        cp16(gb +  8, &lds_b[buf][tid][ 8]);
        cp16(gb + 16, &lds_b[buf][tid][16]);
        cp16(gb + 24, &lds_b[buf][tid][24]);
    };

    const int ksteps = s * 8;                           // (s*256)/32
    issue_tile(0, 0);
    wait_async();
    __syncthreads();

    for (int ks = 0; ks < ksteps; ++ks) {
        const int cur = ks & 1;
        if (ks + 1 < ksteps) issue_tile(ks + 1, cur ^ 1);   // overlap next tile copy

        v16bf afr[2], bfr[4];
        #pragma unroll
        for (int mm = 0; mm < 2; ++mm) {   // A layout: lanes<16 K{0..7,16..23}, lanes>=16 K{8..15,24..31}
            const __bf16* p = &lds_a[cur][wm * 32 + mm * 16 + l16][0];
            union { v16bf v; uint4 q[2]; } u;
            u.q[0] = *reinterpret_cast<const uint4*>(p + half * 8);
            u.q[1] = *reinterpret_cast<const uint4*>(p + 16 + half * 8);
            afr[mm] = u.v;
        }
        #pragma unroll
        for (int nn = 0; nn < 4; ++nn) {   // B layout: lanes<16 K{0..7,16..23}, lanes>=16 K{8..15,24..31}
            const __bf16* p = &lds_b[cur][wn * 64 + nn * 16 + l16][0];
            union { v16bf v; uint4 q[2]; } u;
            u.q[0] = *reinterpret_cast<const uint4*>(p + half * 8);
            u.q[1] = *reinterpret_cast<const uint4*>(p + 16 + half * 8);
            bfr[nn] = u.v;
        }
        #pragma unroll
        for (int mm = 0; mm < 2; ++mm)
            #pragma unroll
            for (int nn = 0; nn < 4; ++nn)
                acc[mm][nn] = __builtin_amdgcn_wmma_f32_16x16x32_bf16(
                    false, afr[mm], false, bfr[nn], (short)0, acc[mm][nn],
                    false, false);

        wait_async();       // this wave's async copy of tile ks+1 complete
        __syncthreads();    // all waves: tile ks+1 visible, tile ks reads done
    }

    // ---- stage 1 epilogue: +b1, relu, bf16 -> LDS H tile ----
    #pragma unroll
    for (int nn = 0; nn < 4; ++nn) {
        const int k = wn * 64 + nn * 16 + l16;
        const float bias = b1[isc * KDIM + k];
        #pragma unroll
        for (int mm = 0; mm < 2; ++mm) {
            const int mbase = wm * 32 + mm * 16 + half * 8;
            #pragma unroll
            for (int v = 0; v < 8; ++v) {
                float h = fmaxf(acc[mm][nn][v] + bias, 0.f);
                lds_h[mbase + v][k] = (__bf16)h;
            }
        }
    }
    __syncthreads();

    // ---- stage 2: wave (wm4 in 0..3 over M, wn2 in 0..1 over C) ----
    const int wm4 = w & 3;
    const int wn2 = w >> 2;

    v8f acc2[6];
    #pragma unroll
    for (int t = 0; t < 6; ++t)
        #pragma unroll
        for (int e = 0; e < 8; ++e) acc2[t][e] = 0.f;

    for (int ks = 0; ks < 8; ++ks) {
        const int kk = ks * 32;
        if (ks) __syncthreads();
        if (tid < 192) {   // W2 tile: 192(c) rows x 32(k)
            const __bf16* src = w2t + (size_t)(isc * CPAD + tid) * KDIM + kk;
            cp16(src +  0, &lds_w2[tid][ 0]);
            cp16(src +  8, &lds_w2[tid][ 8]);
            cp16(src + 16, &lds_w2[tid][16]);
            cp16(src + 24, &lds_w2[tid][24]);
        }
        wait_async();
        __syncthreads();

        v16bf hfr;
        {
            const __bf16* p = &lds_h[wm4 * 16 + l16][kk];
            union { v16bf v; uint4 q[2]; } u;
            u.q[0] = *reinterpret_cast<const uint4*>(p + half * 8);
            u.q[1] = *reinterpret_cast<const uint4*>(p + 16 + half * 8);
            hfr = u.v;
        }
        #pragma unroll
        for (int t = 0; t < 6; ++t) {
            const __bf16* p = &lds_w2[wn2 * 96 + t * 16 + l16][0];
            union { v16bf v; uint4 q[2]; } u;
            u.q[0] = *reinterpret_cast<const uint4*>(p + half * 8);
            u.q[1] = *reinterpret_cast<const uint4*>(p + 16 + half * 8);
            acc2[t] = __builtin_amdgcn_wmma_f32_16x16x32_bf16(
                false, hfr, false, u.v, (short)0, acc2[t], false, false);
        }
    }

    // ---- stage 2 epilogue: accumulate into out (bias pre-seeded) ----
    #pragma unroll
    for (int t = 0; t < 6; ++t) {
        const int c = wn2 * 96 + t * 16 + l16;
        if (c < CDIM) {
            const int mbase = m0 + wm4 * 16 + half * 8;
            #pragma unroll
            for (int v = 0; v < 8; ++v)
                atomicAdd(&out[(size_t)(mbase + v) * CDIM + c], acc2[t][v]);
        }
    }
}

// ---------------- launch ----------------
extern "C" void kernel_launch(void* const* d_in, const int* in_sizes, int n_in,
                              void* d_out, int out_size, void* d_ws, size_t ws_size,
                              hipStream_t stream) {
    const float* x  = (const float*)d_in[0];   // [B, T, D]
    const float* W1 = (const float*)d_in[1];   // [7, 2048, 256]
    const float* b1 = (const float*)d_in[2];   // [7, 256]
    const float* W2 = (const float*)d_in[3];   // [7, 256, 174]
    const float* b2 = (const float*)d_in[4];   // [7, 174]
    float* out = (float*)d_out;                // [B, 174]

    char* ws = (char*)d_ws;
    const size_t xr_bytes  = (size_t)BATCH * TFR * DDIM * 2;          // 64 MB
    const size_t w1t_bytes = (size_t)NSCALE * KDIM * FMAX * 2;        // 7 MB
    __bf16* xr  = (__bf16*)ws;
    __bf16* w1t = (__bf16*)(ws + xr_bytes);
    __bf16* w2t = (__bf16*)(ws + xr_bytes + w1t_bytes);

    const int n4   = BATCH * TFR * DDIM / 4;
    const int n_w1 = NSCALE * KDIM * FMAX;
    const int n_w2 = NSCALE * CPAD * KDIM;
    const int n_o  = BATCH * CDIM;

    k_relu_cvt_x<<<(n4 + 255) / 256, 256, 0, stream>>>(x, xr, n4);
    k_cvt_w1t<<<(n_w1 + 255) / 256, 256, 0, stream>>>(W1, w1t, n_w1);
    k_cvt_w2t<<<(n_w2 + 255) / 256, 256, 0, stream>>>(W2, w2t, n_w2);
    k_init_out<<<(n_o + 255) / 256, 256, 0, stream>>>(b2, out, n_o);

    dim3 grid(BATCH / 64, NREL);
    k_trn_fused<<<grid, 256, 0, stream>>>(xr, w1t, w2t, b1, out);
}
